// TabNetEncoder_32461362823436
// MI455X (gfx1250) — matrix operand order repaired and checked by reading, not compiled, not run on hardware
//
#include <hip/hip_runtime.h>

// ---------------------------------------------------------------------------
// TabNet encoder forward for MI455X (gfx1250, wave32, WMMA).
// GEMMs in bf16 via v_wmma_f32_16x16x32_bf16, f32 accumulate.
// A/B tiles staged with GLOBAL_LOAD_ASYNC_TO_LDS_B128 (ASYNCcnt) and
// double-buffered LDS so chunk k+1 loads overlap chunk k WMMAs; B fragments
// additionally double-buffered in registers so ds-load latency overlaps WMMA.
// Ghost-BN (VBS=128) fused into GEMM epilogue (block = 128 rows x 256 cols),
// stats via ds_add_f32 LDS atomics. GLU pairing done in-register by mapping
// each wave's tiles to column pairs (c, c+128).
// ---------------------------------------------------------------------------

typedef __bf16 bf16;
typedef __attribute__((ext_vector_type(16))) __bf16 v16bf;
typedef __attribute__((ext_vector_type(8)))  float  v8f;

#define NROWS     32768
#define RSQRT2    0.70710678118654752f
#define BN_EPS_F  1e-5f

__device__ __forceinline__ bf16 f2bf(float f) {
  unsigned u = __float_as_uint(f);
  u += 0x7FFFu + ((u >> 16) & 1u);          // round to nearest even
  unsigned short h = (unsigned short)(u >> 16);
  union { unsigned short s; bf16 b; } cv; cv.s = h;
  return cv.b;
}

union FragAB { unsigned u[8]; v16bf v; };

__device__ __forceinline__ unsigned lds_addr32(const void* p) {
  // low 32 bits of a generic pointer to __shared__ == LDS byte address
  return (unsigned)(uintptr_t)p;
}

// Async DMA: 16B from global to LDS, per lane. Tracked by ASYNCcnt.
__device__ __forceinline__ void async_copy_b128(unsigned lds_off, const bf16* g) {
  asm volatile("global_load_async_to_lds_b128 %0, %1, off"
               :
               : "v"(lds_off), "v"((unsigned long long)(uintptr_t)g)
               : "memory");
}
__device__ __forceinline__ void wait_async0() {
  asm volatile("s_wait_asynccnt 0" ::: "memory");
}

__device__ __forceinline__ int colbase(int gluMap, int wn, int t) {
  // gluMap: pair tile t with t+4 at +128 columns (value/gate halves).
  return gluMap ? ((t < 4) ? (wn * 64 + t * 16) : (128 + wn * 64 + (t - 4) * 16))
                : (wn * 128 + t * 16);
}

// ---------------------------------------------------------------------------
// Core: C(128x256) = A(128xK) * W(256xK)^T, bf16 inputs, f32 accumulators.
// 512 threads = 16 waves; wave (wm = wid&7, wn = wid>>3) owns rows
// [16*wm,16*wm+16) and 8 column tiles per colbase().
// K consumed in chunks of 32 with double-buffered async LDS staging.
// ---------------------------------------------------------------------------
template <int GLUMAP>
__device__ __forceinline__ void gemm128x256(const bf16* __restrict__ A,
                                            const bf16* __restrict__ W,
                                            int K, int rowBase,
                                            bf16* sA0, bf16* sA1,
                                            bf16* sB0, bf16* sB1,
                                            v8f acc[8]) {
  const int tid  = threadIdx.x;
  const int lane = tid & 31;
  const int wid  = tid >> 5;
  const int wm   = wid & 7;
  const int wn   = wid >> 3;
  const int r16  = lane & 15;
  const int hi   = lane >> 4;

  #pragma unroll
  for (int t = 0; t < 8; ++t)
    #pragma unroll
    for (int e = 0; e < 8; ++e) acc[t][e] = 0.0f;

  // Per-thread staging slots: A chunk 128x32 (1x16B/thread),
  // B chunk 256x32 (2x16B/thread).
  const int rA = (tid * 8) >> 5, cA = (tid * 8) & 31;
  const int rB1 = ((tid + 512) * 8) >> 5, cB1 = ((tid + 512) * 8) & 31;
  const bf16* gA  = A + (size_t)(rowBase + rA) * K + cA;
  const bf16* gB0 = W + (size_t)rA * K + cA;          // rows 0..127
  const bf16* gB1 = W + (size_t)rB1 * K + cB1;        // rows 128..255
  const unsigned dA[2]  = { lds_addr32(sA0 + rA * 40 + cA),
                            lds_addr32(sA1 + rA * 40 + cA) };
  const unsigned dB0[2] = { lds_addr32(sB0 + rA * 40 + cA),
                            lds_addr32(sB1 + rA * 40 + cA) };
  const unsigned dB1[2] = { lds_addr32(sB0 + rB1 * 40 + cB1),
                            lds_addr32(sB1 + rB1 * 40 + cB1) };

  // prefetch chunk 0 into buffer 0
  async_copy_b128(dA[0],  gA);
  async_copy_b128(dB0[0], gB0);
  async_copy_b128(dB1[0], gB1);
  wait_async0();
  __syncthreads();

  const int nch = K >> 5;
  for (int c = 0; c < nch; ++c) {
    const int cur = c & 1;
    if (c + 1 < nch) { // issue next chunk into the other buffer (overlaps WMMA)
      const int nxt = cur ^ 1;
      const int ko = (c + 1) * 32;
      async_copy_b128(dA[nxt],  gA + ko);
      async_copy_b128(dB0[nxt], gB0 + ko);
      async_copy_b128(dB1[nxt], gB1 + ko);
    }
    const bf16* ldsA = cur ? sA1 : sA0;
    const bf16* ldsB = cur ? sB1 : sB0;

    // A fragment: 16x32 bf16; lane holds row r16, K = hi*8 + {2j | 16+2(j-4)}
    FragAB a;
    const bf16* ap = ldsA + (wm * 16 + r16) * 40;
    #pragma unroll
    for (int j = 0; j < 8; ++j) {
      int kk = hi * 8 + ((j < 4) ? 2 * j : 16 + 2 * (j - 4));
      a.u[j] = *(const unsigned*)(ap + kk);
    }

    // B fragment loader: 32x16; lane holds col r16, K = hi*8 + {2v | 16+2(v-4)}
    auto loadB = [&](FragAB& b, int t) {
      const int nc = colbase(GLUMAP, wn, t);
      const bf16* bp = ldsB + (nc + r16) * 40 + hi * 8;
      #pragma unroll
      for (int v = 0; v < 8; ++v) b.u[v] = *(const unsigned*)(bp + ((v < 4) ? 2 * v : 16 + 2 * (v - 4)));
    };

    // Register-level double buffering: load tile t+1 while WMMA consumes t.
    FragAB b0, b1;
    loadB(b0, 0);
    #pragma unroll
    for (int t = 0; t < 8; ++t) {
      FragAB& curB = (t & 1) ? b1 : b0;
      FragAB& nxtB = (t & 1) ? b0 : b1;
      if (t + 1 < 8) loadB(nxtB, t + 1);
      acc[t] = __builtin_amdgcn_wmma_f32_16x16x32_bf16(
          false, a.v, false, curB.v, (short)0, acc[t], false, false);
    }
    wait_async0();     // next chunk landed (usually already done)
    __syncthreads();   // everyone finished reading current buffer
  }
}

// Ghost-BN stats over the block's 128 rows via LDS float atomics; finalizes
// in place so ssum[c] = scale(c), ssq[c] = shift(c):  hn = v*ssum[c]+ssq[c].
template <int GLUMAP>
__device__ __forceinline__ void gbn_stats(const v8f acc[8],
                                          const float* __restrict__ gam,
                                          const float* __restrict__ bet,
                                          float* ssum, float* ssq) {
  const int tid = threadIdx.x, lane = tid & 31, wid = tid >> 5;
  const int wn = wid >> 3, c16 = lane & 15;
  if (tid < 256) { ssum[tid] = 0.0f; ssq[tid] = 0.0f; }
  __syncthreads();
  #pragma unroll
  for (int t = 0; t < 8; ++t) {
    float s = 0.f, q = 0.f;
    #pragma unroll
    for (int j = 0; j < 8; ++j) { float v = acc[t][j]; s += v; q += v * v; }
    s += __shfl_xor(s, 16, 32);   // combine the two 8-row halves (same col)
    q += __shfl_xor(q, 16, 32);
    if (lane < 16) {
      int c = colbase(GLUMAP, wn, t) + c16;
      atomicAdd(&ssum[c], s);
      atomicAdd(&ssq[c], q);
    }
  }
  __syncthreads();
  if (tid < 256) {
    float s = ssum[tid], q = ssq[tid];
    float mean = s * (1.0f / 128.0f);
    float var  = q * (1.0f / 128.0f) - mean * mean;
    float g    = gam[tid] * rsqrtf(var + BN_EPS_F);
    ssum[tid] = g;                    // scale
    ssq [tid] = bet[tid] - mean * g;  // shift
  }
  __syncthreads();
}

// ---------------------------------------------------------------------------
// Fused GLU layer: out = [resid +] GLU(GBN(A @ W^T)) [* scale]
// Writes f32 (residual chain) and bf16 (next GEMM A operand).
// ---------------------------------------------------------------------------
__global__ __launch_bounds__(512) void glu_kernel(
    const bf16* __restrict__ A, const bf16* __restrict__ W,
    const float* __restrict__ gam, const float* __restrict__ bet,
    const float* __restrict__ resid,
    float* __restrict__ outF, bf16* __restrict__ outB,
    int K, float scale) {
  __shared__ bf16  sA[2][128 * 40];
  __shared__ bf16  sB[2][256 * 40];
  __shared__ float ssum[256], ssq[256];
  const int rowBase = blockIdx.x * 128;

  v8f acc[8];
  gemm128x256<1>(A, W, K, rowBase, sA[0], sA[1], sB[0], sB[1], acc);
  gbn_stats<1>(acc, gam, bet, ssum, ssq);

  const int tid = threadIdx.x, lane = tid & 31, wid = tid >> 5;
  const int wm = wid & 7, wn = wid >> 3, c16 = lane & 15, hi = lane >> 4;
  #pragma unroll
  for (int t = 0; t < 4; ++t) {
    const int colV = wn * 64 + t * 16 + c16;     // value column
    const int colG = colV + 128;                  // paired gate column
    const float sV = ssum[colV], hV = ssq[colV];
    const float sG = ssum[colG], hG = ssq[colG];
    #pragma unroll
    for (int j = 0; j < 8; ++j) {
      const int row = rowBase + wm * 16 + hi * 8 + j;
      float val  = acc[t][j]     * sV + hV;
      float gate = acc[t + 4][j] * sG + hG;
      float hv = val * (1.0f / (1.0f + __expf(-gate)));
      if (resid) hv = (resid[(size_t)row * 128 + colV] + hv) * scale;
      outF[(size_t)row * 128 + colV] = hv;
      outB[(size_t)row * 128 + colV] = f2bf(hv);
    }
  }
}

// ---------------------------------------------------------------------------
// Attention GEMM: a = GBN(att @ att_w^T), K=64, writes normalized a (f32).
// ---------------------------------------------------------------------------
__global__ __launch_bounds__(512) void attgemm_kernel(
    const bf16* __restrict__ A, const bf16* __restrict__ W,
    const float* __restrict__ gam, const float* __restrict__ bet,
    float* __restrict__ aOut) {
  __shared__ bf16  sA[2][128 * 40];
  __shared__ bf16  sB[2][256 * 40];
  __shared__ float ssum[256], ssq[256];
  const int rowBase = blockIdx.x * 128;

  v8f acc[8];
  gemm128x256<0>(A, W, 64, rowBase, sA[0], sA[1], sB[0], sB[1], acc);
  gbn_stats<0>(acc, gam, bet, ssum, ssq);

  const int tid = threadIdx.x, lane = tid & 31, wid = tid >> 5;
  const int wm = wid & 7, wn = wid >> 3, c16 = lane & 15, hi = lane >> 4;
  #pragma unroll
  for (int t = 0; t < 8; ++t) {
    const int col = wn * 128 + t * 16 + c16;
    const float s = ssum[col], h0 = ssq[col];
    #pragma unroll
    for (int j = 0; j < 8; ++j) {
      const int row = rowBase + wm * 16 + hi * 8 + j;
      aOut[(size_t)row * 256 + col] = acc[t][j] * s + h0;
    }
  }
}

// ---------------------------------------------------------------------------
// Sparsemax + prior update + M_loss + masked input (one wave per row).
// tau solved exactly by bisection on f(tau)=sum(max(z-tau,0))-1 (monotone,
// piecewise linear, root in [max(z)-1, max(z)]).
// ---------------------------------------------------------------------------
__global__ __launch_bounds__(512) void sparsemax_kernel(
    const float* __restrict__ a, float* __restrict__ prior,
    const float* __restrict__ xbn, bf16* __restrict__ masked,
    float* __restrict__ mloss) {
  const int lane = threadIdx.x & 31;
  const int wid  = threadIdx.x >> 5;
  const int row  = blockIdx.x * 16 + wid;

  float z[8], pr[8];
  #pragma unroll
  for (int i = 0; i < 8; ++i) {
    int c = lane + 32 * i;
    pr[i] = prior[(size_t)row * 256 + c];
    z[i]  = a[(size_t)row * 256 + c] * pr[i];
  }
  float zmax = z[0];
  #pragma unroll
  for (int i = 1; i < 8; ++i) zmax = fmaxf(zmax, z[i]);
  #pragma unroll
  for (int off = 16; off > 0; off >>= 1)
    zmax = fmaxf(zmax, __shfl_xor(zmax, off, 32));

  float lo = zmax - 1.0f, hiT = zmax;
  for (int it = 0; it < 26; ++it) {
    float tau = 0.5f * (lo + hiT);
    float s = 0.f;
    #pragma unroll
    for (int i = 0; i < 8; ++i) s += fmaxf(z[i] - tau, 0.0f);
    #pragma unroll
    for (int off = 16; off > 0; off >>= 1) s += __shfl_xor(s, off, 32);
    if (s > 1.0f) lo = tau; else hiT = tau;
  }
  const float tau = 0.5f * (lo + hiT);

  float ml = 0.f;
  #pragma unroll
  for (int i = 0; i < 8; ++i) {
    int c = lane + 32 * i;
    float m = fmaxf(z[i] - tau, 0.0f);
    ml += m * logf(m + 1e-15f);
    prior[(size_t)row * 256 + c] = (1.3f - m) * pr[i];
    masked[(size_t)row * 256 + c] = f2bf(m * xbn[(size_t)row * 256 + c]);
  }
  #pragma unroll
  for (int off = 16; off > 0; off >>= 1) ml += __shfl_xor(ml, off, 32);
  if (lane == 0)
    atomicAdd(mloss, ml * (1.0f / (32768.0f * 5.0f)));
}

// ---------------------------------------------------------------------------
// Utility kernels
// ---------------------------------------------------------------------------
__global__ __launch_bounds__(512) void init_kernel(float* __restrict__ prior,
                                                   float* __restrict__ sums,
                                                   float* __restrict__ sqs,
                                                   float* __restrict__ mloss) {
  size_t i = (size_t)blockIdx.x * 512 + threadIdx.x;
  prior[i] = 1.0f;                         // grid sized exactly B*256
  if (i < 256) { sums[i] = 0.0f; sqs[i] = 0.0f; }
  if (i == 0) *mloss = 0.0f;
}

__global__ __launch_bounds__(256) void cvt_kernel(const float* __restrict__ s,
                                                  bf16* __restrict__ d, int n) {
  int i = blockIdx.x * 256 + threadIdx.x;
  if (i < n) d[i] = f2bf(s[i]);
}

__global__ __launch_bounds__(256) void bnstats_kernel(const float* __restrict__ x,
                                                      float* __restrict__ sums,
                                                      float* __restrict__ sqs) {
  const int c = threadIdx.x;
  const int r0 = blockIdx.x * 128;
  float s = 0.f, q = 0.f;
  for (int r = 0; r < 128; ++r) {
    float v = x[(size_t)(r0 + r) * 256 + c];
    s += v; q += v * v;
  }
  atomicAdd(&sums[c], s);
  atomicAdd(&sqs[c], q);
}

__global__ __launch_bounds__(512) void bnnorm_kernel(
    const float* __restrict__ x, const float* __restrict__ g,
    const float* __restrict__ b, const float* __restrict__ sums,
    const float* __restrict__ sqs, float* __restrict__ xbn,
    bf16* __restrict__ abuf) {
  size_t i = (size_t)blockIdx.x * 512 + threadIdx.x;
  int c = (int)(i & 255);
  float mean = sums[c] * (1.0f / 32768.0f);
  float var  = sqs[c] * (1.0f / 32768.0f) - mean * mean;
  float v = (x[i] - mean) * rsqrtf(var + BN_EPS_F) * g[c] + b[c];
  xbn[i]  = v;
  abuf[i] = f2bf(v);
}

// Split transformer output: d_out[step] = relu(h[:, :64]); att = bf16(h[:, 64:])
__global__ __launch_bounds__(512) void split_kernel(const float* __restrict__ h,
                                                    float* __restrict__ outD,
                                                    bf16* __restrict__ attbf,
                                                    int writeOut) {
  size_t i = (size_t)blockIdx.x * 512 + threadIdx.x;
  int row = (int)(i >> 7);
  int c   = (int)(i & 127);
  float v = h[i];
  if (c < 64) {
    if (writeOut) outD[(size_t)row * 64 + c] = fmaxf(v, 0.0f);
  } else {
    attbf[(size_t)row * 64 + (c - 64)] = f2bf(v);
  }
}

// ---------------------------------------------------------------------------
// Host orchestration.
// Input flattening (setup_inputs insertion-order DFS):
//  0: x (B*256)   1: bn_g   2: bn_b   3: shared_fc0 (256x256)  4: shared_fc1
//  5 + ft*10 + {0:sg0 1:sg1 2:sb0 3:sb1 4:iw0 5:iw1 6:ig0 7:ig1 8:ib0 9:ib1}
//  65..69: att_w  70..74: att_g  75..79: att_b
// ---------------------------------------------------------------------------
extern "C" void kernel_launch(void* const* d_in, const int* in_sizes, int n_in,
                              void* d_out, int out_size, void* d_ws, size_t ws_size,
                              hipStream_t stream) {
  (void)in_sizes; (void)n_in; (void)out_size; (void)ws_size;

  const float* x    = (const float*)d_in[0];
  const float* bn_g = (const float*)d_in[1];
  const float* bn_b = (const float*)d_in[2];

  // workspace bump allocator
  char* ws = (char*)d_ws;
  auto alloc = [&](size_t bytes) -> char* {
    char* p = ws;
    ws += (bytes + 255) & ~(size_t)255;
    return p;
  };
  float* xbn   = (float*)alloc((size_t)NROWS * 256 * 4);
  float* prior = (float*)alloc((size_t)NROWS * 256 * 4);
  float* anorm = (float*)alloc((size_t)NROWS * 256 * 4);
  float* hbuf  = (float*)alloc((size_t)NROWS * 128 * 4);
  float* bnsum = (float*)alloc(256 * 4);
  float* bnsq  = (float*)alloc(256 * 4);
  bf16*  abuf  = (bf16*)alloc((size_t)NROWS * 256 * 2);
  bf16*  hbf   = (bf16*)alloc((size_t)NROWS * 128 * 2);
  bf16*  attbf = (bf16*)alloc((size_t)NROWS * 64 * 2);
  bf16*  wfc0  = (bf16*)alloc(65536 * 2);
  bf16*  wfc1  = (bf16*)alloc(32768 * 2);
  bf16*  wiw[12];
  for (int i = 0; i < 12; ++i) wiw[i] = (bf16*)alloc(32768 * 2);
  bf16*  wattw[5];
  for (int s = 0; s < 5; ++s) wattw[s] = (bf16*)alloc(16384 * 2);

  float* outF  = (float*)d_out;
  float* mloss = outF + (size_t)5 * NROWS * 64;

  init_kernel<<<16384, 512, 0, stream>>>(prior, bnsum, bnsq, mloss);

  // weights -> bf16
  cvt_kernel<<<65536 / 256, 256, 0, stream>>>((const float*)d_in[3], wfc0, 65536);
  cvt_kernel<<<32768 / 256, 256, 0, stream>>>((const float*)d_in[4], wfc1, 32768);
  for (int ft = 0; ft < 6; ++ft)
    for (int j = 0; j < 2; ++j)
      cvt_kernel<<<32768 / 256, 256, 0, stream>>>(
          (const float*)d_in[5 + ft * 10 + 4 + j], wiw[ft * 2 + j], 32768);
  for (int s = 0; s < 5; ++s)
    cvt_kernel<<<16384 / 256, 256, 0, stream>>>(
        (const float*)d_in[65 + s], wattw[s], 16384);

  // batch norm
  bnstats_kernel<<<256, 256, 0, stream>>>(x, bnsum, bnsq);
  bnnorm_kernel<<<16384, 512, 0, stream>>>(x, bn_g, bn_b, bnsum, bnsq, xbn, abuf);

  auto runFT = [&](int ft, const bf16* a0) {
    const int base = 5 + ft * 10;
    const float* sg0 = (const float*)d_in[base + 0];
    const float* sg1 = (const float*)d_in[base + 1];
    const float* sb0 = (const float*)d_in[base + 2];
    const float* sb1 = (const float*)d_in[base + 3];
    const float* ig0 = (const float*)d_in[base + 6];
    const float* ig1 = (const float*)d_in[base + 7];
    const float* ib0 = (const float*)d_in[base + 8];
    const float* ib1 = (const float*)d_in[base + 9];
    glu_kernel<<<256, 512, 0, stream>>>(a0, wfc0, sg0, sb0, nullptr,
                                        hbuf, hbf, 256, 1.0f);
    glu_kernel<<<256, 512, 0, stream>>>(hbf, wfc1, sg1, sb1, hbuf,
                                        hbuf, hbf, 128, RSQRT2);
    glu_kernel<<<256, 512, 0, stream>>>(hbf, wiw[ft * 2 + 0], ig0, ib0, hbuf,
                                        hbuf, hbf, 128, RSQRT2);
    glu_kernel<<<256, 512, 0, stream>>>(hbf, wiw[ft * 2 + 1], ig1, ib1, hbuf,
                                        hbuf, hbf, 128, RSQRT2);
  };

  // step-0 transformer on x_bn -> att
  runFT(0, abuf);
  split_kernel<<<8192, 512, 0, stream>>>(hbuf, nullptr, attbf, 0);

  for (int s = 0; s < 5; ++s) {
    const float* ag = (const float*)d_in[70 + s];
    const float* ab = (const float*)d_in[75 + s];
    attgemm_kernel<<<256, 512, 0, stream>>>(attbf, wattw[s], ag, ab, anorm);
    sparsemax_kernel<<<2048, 512, 0, stream>>>(anorm, prior, xbn, abuf, mloss);
    runFT(s + 1, abuf);
    split_kernel<<<8192, 512, 0, stream>>>(
        hbuf, outF + (size_t)s * NROWS * 64, attbf, 1);
  }
}
